// NonLocalBlock_40492951667426
// MI455X (gfx1250) — matrix operand order repaired and checked by reading, not compiled, not run on hardware
//
#include <hip/hip_runtime.h>

// ---------------------------------------------------------------------------
// NonLocalBlock (embedded, sub_sample, residual) for MI455X / gfx1250.
// B=8, H=W=64 (N=4096), C=256, D=128, M=N/2=2048.
//
// Kernel 1: projections theta/phi/g (256->128 GEMMs) on bf16 WMMA,
//           fused stride-2 max-pool; phi stored [m][d], g stored [d][m] so
//           kernel 2's LDS tiles are pure contiguous copies.
// Kernel 2: flash-attention over M=2048 keys with double-buffered
//           GLOBAL_LOAD_ASYNC_TO_LDS staging + fused 128->256 output
//           projection + residual, all on v_wmma_f32_16x16x32_bf16.
// Workspace (d_ws, 16 MB): theta bf16 [8,4096,128], phi bf16 [8,2048,128],
//                          gT bf16 [8,128,2048].
// ---------------------------------------------------------------------------

typedef __attribute__((ext_vector_type(16))) __bf16        v16bf;
typedef __attribute__((ext_vector_type(8)))  float         v8f;
typedef __attribute__((ext_vector_type(4)))  unsigned int  u32x4;
typedef __attribute__((ext_vector_type(4)))  int           i32x4;
typedef unsigned short ushort_t;

union Frag16 {           // one 16x16x32 bf16 WMMA A/B operand (8 VGPRs)
  u32x4   q[2];
  v16bf   v;
  ushort_t s[16];
};

// Native hardware f32->bf16 convert (RNE), not a software bit sequence.
__device__ __forceinline__ ushort_t f32_to_bf16(float f) {
  return __builtin_bit_cast(ushort_t, (__bf16)f);
}
__device__ __forceinline__ unsigned int pack_bf16x2(float lo, float hi) {
  return (unsigned int)f32_to_bf16(lo) | ((unsigned int)f32_to_bf16(hi) << 16);
}

__device__ __forceinline__ float redmax16(float x) {
#pragma unroll
  for (int o = 1; o < 16; o <<= 1) x = fmaxf(x, __shfl_xor(x, o, 32));
  return x;
}
__device__ __forceinline__ float redsum16(float x) {
#pragma unroll
  for (int o = 1; o < 16; o <<= 1) x += __shfl_xor(x, o, 32);
  return x;
}

__device__ __forceinline__ v8f wmma_bf16(v16bf a, v16bf b, v8f c) {
  return __builtin_amdgcn_wmma_f32_16x16x32_bf16(
      /*neg_a=*/false, a, /*neg_b=*/false, b,
      /*c_mod=*/(short)0, c, /*reuse_a=*/false, /*reuse_b=*/false);
}

// ---- CDNA5 async global->LDS DMA (ASYNCcnt), with portable fallback -------
#if __has_builtin(__builtin_amdgcn_global_load_async_to_lds_b128)
#define NLB_ASYNC 1
typedef __attribute__((address_space(1))) i32x4* gvec_t;   // prints as __device__
typedef __attribute__((address_space(3))) i32x4* lvec_t;   // prints as __shared__
__device__ __forceinline__ void copy16_g2lds(const void* g, void* l) {
  __builtin_amdgcn_global_load_async_to_lds_b128(
      (gvec_t)(size_t)g, (lvec_t)(unsigned int)(size_t)l, 0, 0);
}
#else
#define NLB_ASYNC 0
__device__ __forceinline__ void copy16_g2lds(const void* g, void* l) {
  *(u32x4*)l = *(const u32x4*)g;
}
#endif

__device__ __forceinline__ void wait_async_copies() {
#if NLB_ASYNC
#if __has_builtin(__builtin_amdgcn_s_wait_asynccnt)
  __builtin_amdgcn_s_wait_asynccnt(0);
#else
  asm volatile("s_wait_asynccnt 0" ::: "memory");
#endif
#endif
}

#define NB_B 8
#define NB_N 4096
#define NB_M 2048
#define NB_C 256
#define NB_D 128

// ---------------------------------------------------------------------------
// Kernel 1: projections + pooling.
// grid = B*N/64 = 512 blocks, 256 threads (8 wave32).
// Wave wv: row-subtile rt = wv&3 (16 rows), column half ch = wv>>2 (64 cols).
// ---------------------------------------------------------------------------
__global__ __launch_bounds__(256)
void nlb_proj_kernel(const float* __restrict__ x,
                     const float* __restrict__ w_theta,
                     const float* __restrict__ w_phi,
                     const float* __restrict__ w_g,
                     ushort_t* __restrict__ theta_g,   // [B,N,D] bf16
                     ushort_t* __restrict__ phi_g,     // [B,M,D] bf16
                     ushort_t* __restrict__ gT_g) {    // [B,D,M] bf16
  __shared__ __align__(16) ushort_t xs[64 * 256];      // 32 KB x tile (bf16)
  __shared__ __align__(16) ushort_t wct[128 * 32];     // 8 KB  W chunk, [c][k]

  const int t    = threadIdx.x;
  const int lane = t & 31;
  const int wv   = t >> 5;
  const int rt   = wv & 3;
  const int ch   = wv >> 2;

  const int b     = blockIdx.x >> 6;          // 64 blocks per batch
  const int r0loc = (blockIdx.x & 63) * 64;   // first local row
  const size_t R0 = (size_t)b * NB_N + r0loc;

  // ---- stage x tile [64 rows x 256] f32 -> bf16 in LDS (float4 in, u32 out)
  {
    const int row = t >> 2;
    const int c0  = (t & 3) * 64;
    const float* src = x + (R0 + row) * NB_C + c0;
    unsigned int* dst = (unsigned int*)(xs + row * 256 + c0);
#pragma unroll
    for (int i = 0; i < 16; ++i) {
      float4 f = *(const float4*)(src + i * 4);
      dst[i * 2 + 0] = pack_bf16x2(f.x, f.y);
      dst[i * 2 + 1] = pack_bf16x2(f.z, f.w);
    }
  }
  __syncthreads();

  const float* Ws[3] = {w_theta, w_phi, w_g};

  for (int mat = 0; mat < 3; ++mat) {
    const float* W = Ws[mat];
    v8f acc[4];
#pragma unroll
    for (int ct = 0; ct < 4; ++ct) acc[ct] = (v8f){0,0,0,0,0,0,0,0};

    for (int kc = 0; kc < 8; ++kc) {          // K = 256 in chunks of 32
      __syncthreads();
      // stage W chunk transposed: wct[c][k] = bf16(W[kc*32+k][c])
      {
        const int k  = t >> 3;                // 0..31
        const int c0 = (t & 7) * 16;          // 0..112
        const float* wr = W + (size_t)(kc * 32 + k) * NB_D + c0;
#pragma unroll
        for (int i = 0; i < 16; ++i)
          wct[(c0 + i) * 32 + k] = f32_to_bf16(wr[i]);
      }
      __syncthreads();

      // A fragment (x rows, K chunk) from LDS: two contiguous 16B runs
      Frag16 a;
      {
        const int row = rt * 16 + (lane & 15);
        const int kb  = kc * 32 + ((lane < 16) ? 0 : 8);
        a.q[0] = *(const u32x4*)(xs + row * 256 + kb);
        a.q[1] = *(const u32x4*)(xs + row * 256 + kb + 16);
      }
#pragma unroll
      for (int ct = 0; ct < 4; ++ct) {
        Frag16 bf;
        const int c   = ch * 64 + ct * 16 + (lane & 15);
        const int kb2 = (lane < 16) ? 0 : 8;
        bf.q[0] = *(const u32x4*)(wct + c * 32 + kb2);
        bf.q[1] = *(const u32x4*)(wct + c * 32 + kb2 + 16);
        acc[ct] = wmma_bf16(a.v, bf.v, acc[ct]);
      }
    }

    // ---- writeback (C-fragment layout: row = v + (lane>=16?8:0), col = lane&15)
    if (mat == 0) {
#pragma unroll
      for (int ct = 0; ct < 4; ++ct)
#pragma unroll
        for (int v = 0; v < 8; ++v) {
          const int nl = rt * 16 + v + ((lane < 16) ? 0 : 8);
          const int d  = ch * 64 + ct * 16 + (lane & 15);
          theta_g[((size_t)b * NB_N + r0loc + nl) * NB_D + d] =
              f32_to_bf16(acc[ct][v]);
        }
    } else {
      // stride-2 max-pool over row pairs: pairs (2j,2j+1) are adjacent acc lanes
#pragma unroll
      for (int ct = 0; ct < 4; ++ct)
#pragma unroll
        for (int j = 0; j < 4; ++j) {
          const float pv = fmaxf(acc[ct][2 * j], acc[ct][2 * j + 1]);
          const int ml = rt * 8 + j + ((lane < 16) ? 0 : 4);
          const int m  = (r0loc >> 1) + ml;
          const int d  = ch * 64 + ct * 16 + (lane & 15);
          if (mat == 1)
            phi_g[((size_t)b * NB_M + m) * NB_D + d] = f32_to_bf16(pv);  // [m][d]
          else
            gT_g[((size_t)b * NB_D + d) * NB_M + m] = f32_to_bf16(pv);   // [d][m]
        }
    }
  }
}

// ---------------------------------------------------------------------------
// Kernel 2: flash attention over M + fused output projection + residual.
// grid = B * N/128 = 256 blocks, 256 threads (8 wave32).
// Each wave owns 16 query rows; 64 key tiles of 32; D=128 held in registers.
// LDS (40 KB): loop  -> double-buffered {phi[32m][128d] | gT[128d][32m]} + P
//              epilogue -> Y stage 8x(16x128) | w_out tile [16c][128k]
// Key-tile staging = contiguous 16B global->LDS async DMA, double buffered.
// ---------------------------------------------------------------------------
__global__ __launch_bounds__(256)
void nlb_attn_kernel(const float* __restrict__ x,
                     const float* __restrict__ w_out,
                     const ushort_t* __restrict__ theta_g,
                     const ushort_t* __restrict__ phi_g,   // [B,M,D]
                     const ushort_t* __restrict__ gT_g,    // [B,D,M]
                     float* __restrict__ out) {
  __shared__ __align__(16) ushort_t smem[20480];   // 40 KB

  const int t    = threadIdx.x;
  const int lane = t & 31;
  const int wv   = t >> 5;
  const int b    = blockIdx.x >> 5;
  const int q0   = (blockIdx.x & 31) * 128;

  // buffer bases (halves): buf k -> phi @ k*8192, gT @ k*8192+4096
  ushort_t* s_p = smem + 16384 + wv * 512;   // per-wave P stage [16][32]

  const ushort_t* phi_b = phi_g + (size_t)b * NB_M * NB_D;
  const ushort_t* gT_b  = gT_g + (size_t)b * NB_D * NB_M;

  // issue one key tile's DMA: phi chunk is 8 KB contiguous; gT chunk is
  // 128 rows x 64 B (row stride M halves). 256 threads x 2 x 16 B each.
  auto issue_tile = [&](int mt, int buf) {
    ushort_t* lphi = smem + buf * 8192;
    ushort_t* lg   = smem + buf * 8192 + 4096;
    const ushort_t* gp = phi_b + (size_t)(mt * 32) * NB_D;
    copy16_g2lds(gp + t * 8, lphi + t * 8);
    copy16_g2lds(gp + 2048 + t * 8, lphi + 2048 + t * 8);
    const ushort_t* gg = gT_b + mt * 32;
#pragma unroll
    for (int h = 0; h < 2; ++h) {
      const int j   = t + h * 256;
      const int row = j >> 2;
      const int seg = (j & 3) * 8;
      copy16_g2lds(gg + (size_t)row * NB_M + seg, lg + row * 32 + seg);
    }
  };

  // ---- theta A-fragments for this wave's 16 query rows (all of K=128)
  Frag16 thA[4];
  {
    const int nl = q0 + wv * 16 + (lane & 15);
    const ushort_t* tr = theta_g + ((size_t)b * NB_N + nl) * NB_D;
#pragma unroll
    for (int kc = 0; kc < 4; ++kc) {
      const int kb = kc * 32 + ((lane < 16) ? 0 : 8);
      thA[kc].q[0] = *(const u32x4*)(tr + kb);
      thA[kc].q[1] = *(const u32x4*)(tr + kb + 16);
    }
  }

  float mrun[8], lrun[8];
  v8f acc[8];
#pragma unroll
  for (int v = 0; v < 8; ++v) { mrun[v] = -1e30f; lrun[v] = 0.0f; }
#pragma unroll
  for (int tt = 0; tt < 8; ++tt) acc[tt] = (v8f){0,0,0,0,0,0,0,0};

  issue_tile(0, 0);                            // prologue DMA

  for (int mt = 0; mt < NB_M / 32; ++mt) {
    wait_async_copies();                       // this wave's DMA done
    __syncthreads();                           // everyone's DMA visible
    if (mt + 1 < NB_M / 32) issue_tile(mt + 1, (mt + 1) & 1);

    ushort_t* s_phi = smem + (mt & 1) * 8192;         // [m][d]
    ushort_t* s_g   = smem + (mt & 1) * 8192 + 4096;  // [d][m]

    // ---- S = theta . phi^T for two 16-wide key tiles (K = 128)
    v8f s0 = (v8f){0,0,0,0,0,0,0,0};
    v8f s1 = (v8f){0,0,0,0,0,0,0,0};
#pragma unroll
    for (int kc = 0; kc < 4; ++kc) {
      const int kb  = kc * 32 + ((lane < 16) ? 0 : 8);
      const int mc0 = (lane & 15);
      const int mc1 = 16 + (lane & 15);
      Frag16 b0, b1;
      b0.q[0] = *(const u32x4*)(s_phi + mc0 * 128 + kb);
      b0.q[1] = *(const u32x4*)(s_phi + mc0 * 128 + kb + 16);
      b1.q[0] = *(const u32x4*)(s_phi + mc1 * 128 + kb);
      b1.q[1] = *(const u32x4*)(s_phi + mc1 * 128 + kb + 16);
      s0 = wmma_bf16(thA[kc].v, b0.v, s0);
      s1 = wmma_bf16(thA[kc].v, b1.v, s1);
    }

    // ---- online softmax (row stats across the 16-lane half-groups)
#pragma unroll
    for (int v = 0; v < 8; ++v) {
      float mx = redmax16(fmaxf(s0[v], s1[v]));
      float mn = fmaxf(mrun[v], mx);
      float sc = __expf(mrun[v] - mn);
      mrun[v]  = mn;
      float p0 = __expf(s0[v] - mn);
      float p1 = __expf(s1[v] - mn);
      lrun[v]  = lrun[v] * sc + redsum16(p0 + p1);
#pragma unroll
      for (int tt = 0; tt < 8; ++tt) acc[tt][v] *= sc;
      const int row = v + ((lane < 16) ? 0 : 8);
      s_p[row * 32 + (lane & 15)]      = f32_to_bf16(p0);
      s_p[row * 32 + 16 + (lane & 15)] = f32_to_bf16(p1);
    }

    // ---- P as A-fragment (same-wave LDS ops are in-order via DScnt)
    Frag16 pf;
    {
      const int row = lane & 15;
      const int kb  = (lane < 16) ? 0 : 8;
      pf.q[0] = *(const u32x4*)(s_p + row * 32 + kb);
      pf.q[1] = *(const u32x4*)(s_p + row * 32 + kb + 16);
    }

    // ---- acc += P . g  (8 output d-tiles of 16)
#pragma unroll
    for (int tt = 0; tt < 8; ++tt) {
      Frag16 gb;
      const int dc  = tt * 16 + (lane & 15);
      const int kb2 = (lane < 16) ? 0 : 8;
      gb.q[0] = *(const u32x4*)(s_g + dc * 32 + kb2);
      gb.q[1] = *(const u32x4*)(s_g + dc * 32 + kb2 + 16);
      acc[tt] = wmma_bf16(pf.v, gb.v, acc[tt]);
    }
  }

  // ================= epilogue: y/l, then y @ w_out + x =================
  __syncthreads();

  // normalize and stage y as bf16, per-wave region [16 rows][128 d]
  ushort_t* s_y = smem + wv * 2048;
#pragma unroll
  for (int v = 0; v < 8; ++v) {
    const float inv = 1.0f / lrun[v];
    const int row = v + ((lane < 16) ? 0 : 8);
#pragma unroll
    for (int tt = 0; tt < 8; ++tt)
      s_y[row * 128 + tt * 16 + (lane & 15)] = f32_to_bf16(acc[tt][v] * inv);
  }

  Frag16 yA[4];
  {
    const int row = lane & 15;
#pragma unroll
    for (int kc = 0; kc < 4; ++kc) {
      const int kb = kc * 32 + ((lane < 16) ? 0 : 8);
      yA[kc].q[0] = *(const u32x4*)(s_y + row * 128 + kb);
      yA[kc].q[1] = *(const u32x4*)(s_y + row * 128 + kb + 16);
    }
  }

  ushort_t* s_w = smem + 16384;           // [16 c][128 k]
  for (int ct = 0; ct < NB_C / 16; ++ct) {
    __syncthreads();
    // stage w_out column tile transposed, f32 -> bf16
    {
      const int k  = t >> 1;
      const int c0 = (t & 1) * 8;
      const float* src = w_out + (size_t)k * NB_C + ct * 16 + c0;
#pragma unroll
      for (int i = 0; i < 8; ++i)
        s_w[(c0 + i) * 128 + k] = f32_to_bf16(src[i]);
    }
    __syncthreads();

    v8f o = (v8f){0,0,0,0,0,0,0,0};
#pragma unroll
    for (int kc = 0; kc < 4; ++kc) {
      Frag16 wb;
      const int c  = lane & 15;
      const int kb = kc * 32 + ((lane < 16) ? 0 : 8);
      wb.q[0] = *(const u32x4*)(s_w + c * 128 + kb);
      wb.q[1] = *(const u32x4*)(s_w + c * 128 + kb + 16);
      o = wmma_bf16(yA[kc].v, wb.v, o);
    }

    // residual add + store f32
#pragma unroll
    for (int v = 0; v < 8; ++v) {
      const int nl = q0 + wv * 16 + v + ((lane < 16) ? 0 : 8);
      const size_t oi = ((size_t)b * NB_N + nl) * NB_C + ct * 16 + (lane & 15);
      out[oi] = x[oi] + o[v];
    }
  }
}

// ---------------------------------------------------------------------------
extern "C" void kernel_launch(void* const* d_in, const int* in_sizes, int n_in,
                              void* d_out, int out_size, void* d_ws, size_t ws_size,
                              hipStream_t stream) {
  const float* x       = (const float*)d_in[0];
  const float* w_theta = (const float*)d_in[1];
  const float* w_phi   = (const float*)d_in[2];
  const float* w_g     = (const float*)d_in[3];
  const float* w_out   = (const float*)d_in[4];
  float* out = (float*)d_out;

  // workspace layout (bf16): theta | phi | gT  => 16 MB total
  ushort_t* theta = (ushort_t*)d_ws;
  ushort_t* phi   = theta + (size_t)NB_B * NB_N * NB_D;
  ushort_t* gT    = phi   + (size_t)NB_B * NB_M * NB_D;

  nlb_proj_kernel<<<NB_B * NB_N / 64, 256, 0, stream>>>(
      x, w_theta, w_phi, w_g, theta, phi, gT);

  nlb_attn_kernel<<<NB_B * (NB_N / 128), 256, 0, stream>>>(
      x, w_out, theta, phi, gT, out);
}
